// encoder_40218073759787
// MI455X (gfx1250) — matrix operand order repaired and checked by reading, not compiled, not run on hardware
//
#include <hip/hip_runtime.h>
#include <hip/hip_bf16.h>

#define NN 8192
#define FF 1024

typedef __bf16 v16bf __attribute__((ext_vector_type(16)));
typedef float  v8f   __attribute__((ext_vector_type(8)));

union BF16x16 { v16bf v; uint4 q[2]; };

__device__ __forceinline__ __bf16 f2bf(float f) {
    unsigned u = __builtin_bit_cast(unsigned, f);
    unsigned r = (u + 0x7FFFu + ((u >> 16) & 1u)) >> 16;   // RNE
    return __builtin_bit_cast(__bf16, (unsigned short)r);
}
__device__ __forceinline__ float bf2f(__bf16 b) {
    unsigned u = ((unsigned)__builtin_bit_cast(unsigned short, b)) << 16;
    return __builtin_bit_cast(float, u);
}
__device__ __forceinline__ unsigned pk2bf(float lo, float hi) {
#if __has_builtin(__builtin_amdgcn_cvt_pk_bf16_f32)
    auto p = __builtin_amdgcn_cvt_pk_bf16_f32(lo, hi);
    return __builtin_bit_cast(unsigned, p);
#else
    return (unsigned)__builtin_bit_cast(unsigned short, f2bf(lo)) |
           ((unsigned)__builtin_bit_cast(unsigned short, f2bf(hi)) << 16);
#endif
}
// exp(sigmoid(m)): 2 trans ops via v_tanh_f32 when available, else 3.
__device__ __forceinline__ float exp_sigmoid(float m) {
#if __has_builtin(__builtin_amdgcn_tanhf)
    float s = __builtin_fmaf(0.5f, __builtin_amdgcn_tanhf(0.5f * m), 0.5f);
    return __expf(s);
#else
    return __expf(1.f / (1.f + __expf(-m)));
#endif
}

// ---------------------------------------------------------------------------
// fp32 -> bf16 one-shot conversion (X, W matrices).
// ---------------------------------------------------------------------------
__global__ __launch_bounds__(256) void convert_bf16_kernel(
    const float* __restrict__ src, __bf16* __restrict__ dst) {
    int i = (blockIdx.x * 256 + threadIdx.x) * 4;
    float4 v = *(const float4*)(src + i);
    uint2 pk;
    pk.x = pk2bf(v.x, v.y);
    pk.y = pk2bf(v.z, v.w);
    *(uint2*)(dst + i) = pk;
}

// ---------------------------------------------------------------------------
// Pass 0: A (fp32) -> A_bf16 + AT_bf16 (transposed), both row-major bf16,
// so attention streams contiguously; halves the HBM-bound A traffic.
// ---------------------------------------------------------------------------
__global__ __launch_bounds__(256) void convert_transpose_kernel(
    const float* __restrict__ A, __bf16* __restrict__ Ab, __bf16* __restrict__ ATb) {
    __shared__ __bf16 tile[32][34];
    int tid = threadIdx.x;
    int cx = (tid & 15) * 2;
    int ry = tid >> 4;
    int i0 = blockIdx.y * 32, j0 = blockIdx.x * 32;
#pragma unroll
    for (int rr = 0; rr < 2; ++rr) {
        int row = ry + rr * 16;
        float2 v = *(const float2*)(A + (size_t)(i0 + row) * NN + j0 + cx);
        unsigned pk = pk2bf(v.x, v.y);
        *(unsigned*)(Ab + (size_t)(i0 + row) * NN + j0 + cx) = pk;
        tile[row][cx]     = __builtin_bit_cast(__bf16, (unsigned short)(pk & 0xFFFFu));
        tile[row][cx + 1] = __builtin_bit_cast(__bf16, (unsigned short)(pk >> 16));
    }
    __syncthreads();
#pragma unroll
    for (int rr = 0; rr < 2; ++rr) {
        int row = ry + rr * 16;
        unsigned s0 = (unsigned)__builtin_bit_cast(unsigned short, tile[cx][row]);
        unsigned s1 = (unsigned)__builtin_bit_cast(unsigned short, tile[cx + 1][row]);
        *(unsigned*)(ATb + (size_t)(j0 + row) * NN + i0 + cx) = s0 | (s1 << 16);
    }
}

// ---------------------------------------------------------------------------
// Pass 1 per layer: Zt[f][i] = relu(H @ W^T + b); pure-bf16 operand loop.
// ---------------------------------------------------------------------------
template <int IN_F, int OUT_F>
__global__ __launch_bounds__(256) void zgemm_kernel(
    const __bf16* __restrict__ H, const __bf16* __restrict__ W,
    const float* __restrict__ Wb, __bf16* __restrict__ Zt) {
    int tid  = threadIdx.x;
    int wave = tid >> 5, lane = tid & 31, l16 = lane & 15;
    bool hiL = lane >= 16;

    int g  = blockIdx.x * 8 + wave;
    int ft = g % (OUT_F / 16), it = g / (OUT_F / 16);
    int i0 = it * 16, f0 = ft * 16;

    const __bf16* hrow = H + (size_t)(i0 + l16) * IN_F;
    const __bf16* wrow = W + (size_t)(f0 + l16) * IN_F;

    v8f acc = {0.f, 0.f, 0.f, 0.f, 0.f, 0.f, 0.f, 0.f};

#pragma unroll 2
    for (int kk = 0; kk < IN_F / 32; ++kk) {
        const __bf16* ha = hrow + kk * 32 + (hiL ? 8 : 0);   // A: K{0..7,16..23}/{8..15,24..31}
        const __bf16* wb = wrow + kk * 32 + (hiL ? 8 : 0);   // B: K{0..7,16..23}/{8..15,24..31}
        BF16x16 ua, ub;
        ua.q[0] = *(const uint4*)ha;
        ua.q[1] = *(const uint4*)(ha + 16);
        ub.q[0] = *(const uint4*)wb;
        ub.q[1] = *(const uint4*)(wb + 16);
        acc = __builtin_amdgcn_wmma_f32_16x16x32_bf16(false, ua.v, false, ub.v,
                                                      (short)0, acc, false, false);
    }

    float bias = Wb[f0 + l16];
    __bf16* zcol = Zt + (size_t)(f0 + l16) * NN + i0 + (hiL ? 8 : 0);
#pragma unroll
    for (int gg = 0; gg < 8; gg += 2) {   // consecutive gg = consecutive i: pack u32
        float z0 = acc[gg] + bias;     z0 = z0 > 0.f ? z0 : 0.f;
        float z1 = acc[gg + 1] + bias; z1 = z1 > 0.f ? z1 : 0.f;
        *(unsigned*)(zcol + gg) = pk2bf(z0, z1);
    }
}

// ---------------------------------------------------------------------------
// Pass 2 per layer: t[i], r[i] fused GEMVs + biases.
// ---------------------------------------------------------------------------
template <int OUT_F>
__global__ __launch_bounds__(256) void tr_vec_kernel(
    const __bf16* __restrict__ Zt, const float* __restrict__ tw, const float* __restrict__ tb,
    const float* __restrict__ rw, const float* __restrict__ rb,
    float* __restrict__ t, float* __restrict__ r) {
    int i = blockIdx.x * 256 + threadIdx.x;
    float ta = 0.f, ra = 0.f;
    for (int f = 0; f < OUT_F; ++f) {
        float z = bf2f(Zt[(size_t)f * NN + i]);
        ta += z * tw[f];
        ra += z * rw[f];
    }
    t[i] = ta + tb[0];
    r[i] = ra + rb[0];
}

// ---------------------------------------------------------------------------
// Pass 3 per layer: fused flash-style attention, 32 rows per block.
//   P[i][j] = exp(sigmoid(A[i,j]*t[j] + A[j,i]*r[i]))
//   out[i]  = (P @ Z)[i] / sum_j P[i][j]
// Two 16-row WMMA A-tiles per block: B-operand (Zt) registers reused 2x,
// halves Zt L2 traffic. Ping-pong P buffers -> one barrier per j-step.
// ---------------------------------------------------------------------------
template <int OUT_F, int NW, bool BFOUT>
__global__ __launch_bounds__(NW * 32) void attn_kernel(
    const __bf16* __restrict__ Ab, const __bf16* __restrict__ ATb,
    const __bf16* __restrict__ Zt, const float* __restrict__ tvec,
    const float* __restrict__ rvec, void* __restrict__ OutP) {
    constexpr int TPW = OUT_F / 16 / NW;   // f-tiles per wave
    constexpr int NT  = NW * 32;
    constexpr int E   = 1024 / NT;         // P elements per thread (32 rows x 32 cols)
    constexpr int TPR = 32 / E;            // threads per P row

    __shared__ unsigned P_lds[2][2][16 * 16];  // [pingpong][rowtile][row*16+colpair]
    __shared__ float den_s[32];

    int tid = threadIdx.x;
    int wave = tid >> 5, lane = tid & 31, l16 = lane & 15;
    bool hiL = lane >= 16;
    int i0 = blockIdx.x * 32;

    int prow = tid / TPR;                  // 0..31
    int pcol = (tid % TPR) * E;
    float r_i = rvec[i0 + prow];
    float den_local = 0.f;
    if (tid < 32) den_s[tid] = 0.f;

    const __bf16* arow  = Ab  + (size_t)(i0 + prow) * NN;
    const __bf16* atrow = ATb + (size_t)(i0 + prow) * NN;
    unsigned* pst = &P_lds[0][prow >> 4][(prow & 15) * 16];

    v8f acc[TPW][2];
#pragma unroll
    for (int tt = 0; tt < TPW; ++tt)
#pragma unroll
        for (int rt = 0; rt < 2; ++rt)
            acc[tt][rt] = (v8f){0.f, 0.f, 0.f, 0.f, 0.f, 0.f, 0.f, 0.f};

    int p = 0;
    for (int j0 = 0; j0 < NN; j0 += 32, p ^= 1) {
        // ---- phase 1: P tile (32x32) -> LDS[p] ----
        __builtin_prefetch(arow  + j0 + 1024, 0, 1);   // global_prefetch_b8
        __builtin_prefetch(atrow + j0 + 1024, 0, 1);
#pragma unroll
        for (int e = 0; e < E; e += 2) {
            int c = pcol + e;
            unsigned av = *(const unsigned*)(arow + j0 + c);
            unsigned qv = *(const unsigned*)(atrow + j0 + c);
            float2 tv = *(const float2*)(tvec + j0 + c);
            float a0 = bf2f(__builtin_bit_cast(__bf16, (unsigned short)(av & 0xFFFFu)));
            float a1 = bf2f(__builtin_bit_cast(__bf16, (unsigned short)(av >> 16)));
            float q0 = bf2f(__builtin_bit_cast(__bf16, (unsigned short)(qv & 0xFFFFu)));
            float q1 = bf2f(__builtin_bit_cast(__bf16, (unsigned short)(qv >> 16)));
            float p0 = exp_sigmoid(a0 * tv.x + q0 * r_i);
            float p1 = exp_sigmoid(a1 * tv.y + q1 * r_i);
            den_local += p0 + p1;
            pst[p * 512 + (c >> 1)] = pk2bf(p0, p1);
        }
        __syncthreads();   // single barrier: ping-pong makes the tail barrier redundant

        // ---- phase 2: WMMA accumulate; B reused across both row tiles ----
        BF16x16 ua[2];
#pragma unroll
        for (int rt = 0; rt < 2; ++rt) {
            const unsigned* pr = &P_lds[p][rt][l16 * 16];
            ua[rt].q[0] = *(const uint4*)(pr + (hiL ? 4 : 0));
            ua[rt].q[1] = *(const uint4*)(pr + 8 + (hiL ? 4 : 0));
        }
#pragma unroll
        for (int tt = 0; tt < TPW; ++tt) {
            int f = (wave * TPW + tt) * 16 + l16;
            const __bf16* zp = Zt + (size_t)f * NN + j0 + (hiL ? 8 : 0);
            BF16x16 ub;
            ub.q[0] = *(const uint4*)zp;
            ub.q[1] = *(const uint4*)(zp + 16);
#pragma unroll
            for (int rt = 0; rt < 2; ++rt)
                acc[tt][rt] = __builtin_amdgcn_wmma_f32_16x16x32_bf16(
                    false, ua[rt].v, false, ub.v, (short)0, acc[tt][rt], false, false);
        }
    }

    atomicAdd(&den_s[prow], den_local);   // ds_add_f32
    __syncthreads();

#pragma unroll
    for (int tt = 0; tt < TPW; ++tt) {
        int f = (wave * TPW + tt) * 16 + l16;
#pragma unroll
        for (int rt = 0; rt < 2; ++rt)
#pragma unroll
            for (int gg = 0; gg < 8; ++gg) {
                int M = rt * 16 + gg + (hiL ? 8 : 0);
                float o = acc[tt][rt][gg] / den_s[M];
                if (BFOUT)
                    ((__bf16*)OutP)[(size_t)(i0 + M) * OUT_F + f] = f2bf(o);
                else
                    ((float*)OutP)[(size_t)(i0 + M) * OUT_F + f] = o;
            }
    }
}

// ---------------------------------------------------------------------------
// Workspace layout (bytes)
// ---------------------------------------------------------------------------
#define OFF_ABF   ((size_t)0)                       // 8192*8192*2 = 128 MB
#define OFF_ATBF  ((size_t)134217728)               // 128 MB
#define OFF_ZT    ((size_t)268435456)               // 512*8192*2  =   8 MB
#define OFF_T     ((size_t)276824064)               // 8192*4
#define OFF_R     ((size_t)276856832)               // 8192*4
#define OFF_XB    ((size_t)276889600)               // 8192*1024*2 =  16 MB
#define OFF_H1B   ((size_t)293666816)               // 8192*512*2  =   8 MB
#define OFF_H2B   ((size_t)302055424)               // 8192*256*2  =   4 MB
#define OFF_WB    ((size_t)306249728)               // up to 1 MB

extern "C" void kernel_launch(void* const* d_in, const int* in_sizes, int n_in,
                              void* d_out, int out_size, void* d_ws, size_t ws_size,
                              hipStream_t stream) {
    const float* X    = (const float*)d_in[0];
    const float* A    = (const float*)d_in[1];
    const float* W1w  = (const float*)d_in[2];
    const float* W1b  = (const float*)d_in[3];
    const float* t1w  = (const float*)d_in[4];
    const float* t1b  = (const float*)d_in[5];
    const float* r1w  = (const float*)d_in[6];
    const float* r1b  = (const float*)d_in[7];
    const float* W2w  = (const float*)d_in[8];
    const float* W2b  = (const float*)d_in[9];
    const float* t2w  = (const float*)d_in[10];
    const float* t2b  = (const float*)d_in[11];
    const float* r2w  = (const float*)d_in[12];
    const float* r2b  = (const float*)d_in[13];
    const float* W3w  = (const float*)d_in[14];
    const float* W3b  = (const float*)d_in[15];
    const float* t3w  = (const float*)d_in[16];
    const float* t3b  = (const float*)d_in[17];
    const float* r3w  = (const float*)d_in[18];
    const float* r3b  = (const float*)d_in[19];

    char* ws = (char*)d_ws;
    __bf16* Ab  = (__bf16*)(ws + OFF_ABF);
    __bf16* ATb = (__bf16*)(ws + OFF_ATBF);
    __bf16* Zt  = (__bf16*)(ws + OFF_ZT);
    float*  tv  = (float*)(ws + OFF_T);
    float*  rv  = (float*)(ws + OFF_R);
    __bf16* XB  = (__bf16*)(ws + OFF_XB);
    __bf16* H1B = (__bf16*)(ws + OFF_H1B);
    __bf16* H2B = (__bf16*)(ws + OFF_H2B);
    __bf16* WB  = (__bf16*)(ws + OFF_WB);
    float*  H3  = (float*)d_out;

    convert_transpose_kernel<<<dim3(NN / 32, NN / 32), 256, 0, stream>>>(A, Ab, ATb);
    convert_bf16_kernel<<<(NN * FF) / 4 / 256, 256, 0, stream>>>(X, XB);

    // ---- Layer 1: 1024 -> 512 ----
    convert_bf16_kernel<<<(512 * 1024) / 4 / 256, 256, 0, stream>>>(W1w, WB);
    zgemm_kernel<1024, 512><<<(NN / 16) * (512 / 16) / 8, 256, 0, stream>>>(XB, WB, W1b, Zt);
    tr_vec_kernel<512><<<NN / 256, 256, 0, stream>>>(Zt, t1w, t1b, r1w, r1b, tv, rv);
    attn_kernel<512, 8, true><<<NN / 32, 256, 0, stream>>>(Ab, ATb, Zt, tv, rv, H1B);

    // ---- Layer 2: 512 -> 256 ----
    convert_bf16_kernel<<<(256 * 512) / 4 / 256, 256, 0, stream>>>(W2w, WB);
    zgemm_kernel<512, 256><<<(NN / 16) * (256 / 16) / 8, 256, 0, stream>>>(H1B, WB, W2b, Zt);
    tr_vec_kernel<256><<<NN / 256, 256, 0, stream>>>(Zt, t2w, t2b, r2w, r2b, tv, rv);
    attn_kernel<256, 8, true><<<NN / 32, 256, 0, stream>>>(Ab, ATb, Zt, tv, rv, H2B);

    // ---- Layer 3: 256 -> 64 ----
    convert_bf16_kernel<<<(64 * 256) / 4 / 256, 256, 0, stream>>>(W3w, WB);
    zgemm_kernel<256, 64><<<(NN / 16) * (64 / 16) / 8, 256, 0, stream>>>(H2B, WB, W3b, Zt);
    tr_vec_kernel<64><<<NN / 256, 256, 0, stream>>>(Zt, t3w, t3b, r3w, r3b, tv, rv);
    attn_kernel<64, 4, false><<<NN / 32, 128, 0, stream>>>(Ab, ATb, Zt, tv, rv, H3);
}
